// F_PVMLayer_14310831031044
// MI455X (gfx1250) — matrix operand order repaired and checked by reading, not compiled, not run on hardware
//
#include <hip/hip_runtime.h>
#include <math.h>

typedef __attribute__((ext_vector_type(16))) _Float16 v16h;
typedef __attribute__((ext_vector_type(8)))  float    v8f;

#define LSEQ 16384

// ---------------- workspace layout (bytes) ----------------
#define OFF_XNT   ((size_t)0)                     // f32 [4][128][L]      33.5MB
#define OFF_XZT   ((size_t)33554432)              // f16 [16][128][L]     67.1MB
#define OFF_UT    ((size_t)100663296)             // f16 [16][64][L]      33.5MB
#define OFF_DPT   ((size_t)134217728)             // f32 [16][64][L]      67.1MB
#define OFF_BMT   ((size_t)201326592)             // f32 [16][16][L]      16.8MB
#define OFF_CMT   ((size_t)218103808)             // f32 [16][16][L]      16.8MB
#define OFF_APROD ((size_t)234881024)             // f32 [16][128][1024]   8.4MB
#define OFF_HLOC  ((size_t)243269632)             // f32 same              8.4MB
#define OFF_HIN   ((size_t)251658240)             // f32 same              8.4MB
#define OFF_YGT   ((size_t)260046848)             // f16 [16][64][L]      33.5MB
#define OFF_YCATT ((size_t)293601280)             // f32 [4][128][L]      33.5MB
#define OFF_XMT   ((size_t)327155712)             // f16 [4][128][L]      16.8MB
#define OFF_B1    ((size_t)343932928)             // f16 swizzled, 8 tiles  (8KB)
#define OFF_B2    ((size_t)343941120)             // f16 swizzled, 12 tiles (12KB)
#define OFF_B3    ((size_t)343953408)             // f16 swizzled, 4 tiles  (4KB)
#define OFF_B4    ((size_t)343957504)             // f16 swizzled, 32 tiles (32KB)
#define OFF_BCAT  ((size_t)343990272)             // f32 [128]
#define OFF_ANEG  ((size_t)343990784)             // f32 [64][16]

// ---------------- WMMA fragment helpers (CDNA5 wave32 layouts) ----------------
__device__ inline v16h load_a_frag(const _Float16* src, int ld) {
  // A 16x32 f16, row-major src[m][k]
  int lane = threadIdx.x & 31;
  int m = lane & 15;
  int kb = (lane >> 4) << 4;   // 0 or 16
  v16h a;
#pragma unroll
  for (int i = 0; i < 8; ++i)  a[i]     = src[m * ld + kb + i];
#pragma unroll
  for (int i = 0; i < 8; ++i)  a[8 + i] = src[m * ld + kb + 8 + i];
  return a;
}

// B fragments are pre-swizzled by k_prep into per-lane-contiguous order:
// tile index -> 32 lanes x 16 halves (32B per lane) -> one v16h load per lane.
__device__ inline v16h load_b_swz(const _Float16* base, int tile) {
  int lane = threadIdx.x & 31;
  return ((const v16h*)base)[tile * 32 + lane];
}

__device__ inline float softplus_f(float x) {
  return (x > 20.f) ? x : log1pf(expf(x));
}
__device__ inline float silu_f(float x) { return x / (1.f + expf(-x)); }

// ---------------- K0: weight prep (build swizzled f16 B fragments) ----------------
__global__ void k_prep(const float* Win, const float* Wx, const float* Wdt, const float* A_log,
                       const float* W1, const float* b1, const float* W2, const float* b2,
                       const float* W3, const float* b3, const float* W4, const float* b4,
                       const float* W5, const float* b5, const float* Wout,
                       _Float16* B1, _Float16* B2, _Float16* B3, _Float16* B4,
                       float* bcat, float* Aneg) {
  int t = threadIdx.x;
  // B1: K=32, N=128 -> 8 n-tiles. B[k][n] = Win[n][k]
  for (int e = t; e < 8 * 512; e += 256) {
    int i = e & 15, lane = (e >> 4) & 31, ntile = e >> 9;
    int n = ntile * 16 + (lane & 15);
    int k = ((lane >> 4) << 4) + i;
    B1[e] = (_Float16)Win[n * 32 + k];
  }
  // B2: K=64, N=96 -> tiles = ks*6 + ntile. cols: [0,64)=Wcomb, [64,80)=WxB, [80,96)=WxC
  for (int e = t; e < 12 * 512; e += 256) {
    int i = e & 15, lane = (e >> 4) & 31, tile = e >> 9;
    int ntile = tile % 6, ks = tile / 6;
    int n = ntile * 16 + (lane & 15);
    int k = ks * 32 + ((lane >> 4) << 4) + i;
    float v;
    if (n < 64)      v = Wx[k] * Wdt[n * 2 + 0] + Wx[64 + k] * Wdt[n * 2 + 1];
    else if (n < 80) v = Wx[(2 + (n - 64)) * 64 + k];
    else             v = Wx[(18 + (n - 80)) * 64 + k];
    B2[e] = (_Float16)v;
  }
  // B3: K=64, N=32 -> tiles = ks*2 + ntile. B[k][n] = Wout[n][k]
  for (int e = t; e < 4 * 512; e += 256) {
    int i = e & 15, lane = (e >> 4) & 31, tile = e >> 9;
    int ntile = tile & 1, ks = tile >> 1;
    int n = ntile * 16 + (lane & 15);
    int k = ks * 32 + ((lane >> 4) << 4) + i;
    B3[e] = (_Float16)Wout[n * 64 + k];
  }
  // B4: K=128, N=128 -> tiles = ks*8 + ntile. columns = concat(W1..W5) rows
  for (int e = t; e < 32 * 512; e += 256) {
    int i = e & 15, lane = (e >> 4) & 31, tile = e >> 9;
    int ntile = tile & 7, ks = tile >> 3;
    int n = ntile * 16 + (lane & 15);
    int k = ks * 32 + ((lane >> 4) << 4) + i;
    const float* Wp; int cp;
    if (n < 8)       { Wp = W1; cp = n; }
    else if (n < 24) { Wp = W2; cp = n - 8; }
    else if (n < 48) { Wp = W3; cp = n - 24; }
    else if (n < 80) { Wp = W4; cp = n - 48; }
    else             { Wp = W5; cp = n - 80; }
    B4[e] = (_Float16)Wp[cp * 128 + k];
  }
  if (t < 128) {
    int n = t; const float* bp; int cp;
    if (n < 8)       { bp = b1; cp = n; }
    else if (n < 24) { bp = b2; cp = n - 8; }
    else if (n < 48) { bp = b3; cp = n - 24; }
    else if (n < 80) { bp = b4; cp = n - 48; }
    else             { bp = b5; cp = n - 80; }
    bcat[n] = bp[cp];
  }
  for (int e = t; e < 1024; e += 256) Aneg[e] = -expf(A_log[e]);
}

// ---------------- K1/K9: layernorm over 128 channels, channel-major src/dst ----------------
template <typename OutT>
__global__ void k_layernorm(const float* __restrict__ src, const float* g, const float* beta,
                            OutT* __restrict__ dst) {
  int row = blockIdx.x * blockDim.x + threadIdx.x;   // 65536 rows = (b,l)
  int b = row >> 14, l = row & (LSEQ - 1);
  const float* p = src + (size_t)b * 128 * LSEQ + l;
  float s = 0.f, s2 = 0.f;
  for (int c = 0; c < 128; ++c) { float v = p[(size_t)c * LSEQ]; s += v; s2 += v * v; }
  float mu = s * (1.f / 128.f);
  float var = s2 * (1.f / 128.f) - mu * mu;
  float rs = rsqrtf(var + 1e-5f);
  OutT* q = dst + (size_t)b * 128 * LSEQ + l;
  for (int c = 0; c < 128; ++c) {
    float v = (p[(size_t)c * LSEQ] - mu) * rs * g[c] + beta[c];
    q[(size_t)c * LSEQ] = (OutT)v;
  }
}

// ---------------- K2: xz = xs @ Win^T   (M=16L, K=32, N=128) ----------------
__global__ void k_gemm_xz(const float* __restrict__ xnT, const _Float16* __restrict__ B1,
                          _Float16* __restrict__ xzT) {
  __shared__ _Float16 At[16 * 32];
  __shared__ _Float16 Ot[128 * 16];
  int bb = blockIdx.y;
  int b = bb & 3, cidx = bb >> 2;
  int l0 = blockIdx.x * 16;
  int t = threadIdx.x;
  for (int e = t; e < 512; e += 256) {
    int k = e >> 4, m = e & 15;
    At[m * 32 + k] = (_Float16)xnT[((size_t)(b * 128 + cidx * 32 + k)) * LSEQ + l0 + m];
  }
  __syncthreads();
  int w = t >> 5;
  int n0 = w * 16;
  v16h a = load_a_frag(At, 32);
  v16h bf = load_b_swz(B1, w);
  v8f c = {};
  c = __builtin_amdgcn_wmma_f32_16x16x32_f16(false, a, false, bf, (short)0, c, false, false);
  { int lane = t & 31; int n = lane & 15, mb = (lane >> 4) << 3;
#pragma unroll
    for (int j = 0; j < 8; ++j) Ot[(n0 + n) * 16 + mb + j] = (_Float16)c[j]; }
  __syncthreads();
  if (t < 128) {
    _Float16* dst = xzT + ((size_t)(bb * 128 + t)) * LSEQ + l0;
    for (int j = 0; j < 16; ++j) dst[j] = Ot[t * 16 + j];
  }
}

// ---------------- K3: causal depthwise conv4 + SiLU ----------------
__global__ void k_conv(const _Float16* __restrict__ xzT, const float* Wconv, const float* bconv,
                       _Float16* __restrict__ uT) {
  size_t idx = (size_t)blockIdx.x * blockDim.x + threadIdx.x;  // 16*64*L
  int l = (int)(idx & (LSEQ - 1));
  int d = (int)((idx >> 14) & 63);
  int bb = (int)(idx >> 20);
  const _Float16* xr = xzT + ((size_t)(bb * 128 + d)) * LSEQ;
  float acc = bconv[d];
#pragma unroll
  for (int j = 0; j < 4; ++j) {
    int li = l - 3 + j;
    float xv = (li >= 0) ? (float)xr[li] : 0.f;
    acc += Wconv[d * 4 + j] * xv;
  }
  uT[((size_t)(bb * 64 + d)) * LSEQ + l] = (_Float16)silu_f(acc);
}

// ---------------- K4: xdbl = u @ [Wcomb | WxB | WxC]  (K=64, N=96) ----------------
__global__ void k_gemm_xdbl(const _Float16* __restrict__ uT, const _Float16* __restrict__ B2,
                            const float* bdt, float* __restrict__ dpT,
                            float* __restrict__ BmT, float* __restrict__ CmT) {
  __shared__ _Float16 At[16 * 64];
  __shared__ float Ot[96 * 16];
  int bb = blockIdx.y;
  int l0 = blockIdx.x * 16;
  int t = threadIdx.x;  // 192 threads = 6 waves
  for (int e = t; e < 1024; e += 192) {
    int k = e >> 4, m = e & 15;
    At[m * 64 + k] = uT[((size_t)(bb * 64 + k)) * LSEQ + l0 + m];
  }
  __syncthreads();
  int w = t >> 5;
  int n0 = w * 16;
  v8f c = {};
#pragma unroll
  for (int ks = 0; ks < 2; ++ks) {
    v16h a = load_a_frag(At + ks * 32, 64);
    v16h bf = load_b_swz(B2, ks * 6 + w);
    c = __builtin_amdgcn_wmma_f32_16x16x32_f16(false, a, false, bf, (short)0, c, false, false);
  }
  { int lane = t & 31; int n = lane & 15, mb = (lane >> 4) << 3;
    float bias = (n0 < 64) ? bdt[n0 + n] : 0.f;
#pragma unroll
    for (int j = 0; j < 8; ++j) Ot[(n0 + n) * 16 + mb + j] = c[j] + bias; }
  __syncthreads();
  if (t < 96) {
    const float* src = Ot + t * 16;
    float* dst;
    if (t < 64)      dst = dpT + ((size_t)(bb * 64 + t)) * LSEQ + l0;
    else if (t < 80) dst = BmT + ((size_t)(bb * 16 + (t - 64))) * LSEQ + l0;
    else             dst = CmT + ((size_t)(bb * 16 + (t - 80))) * LSEQ + l0;
    for (int j = 0; j < 16; ++j) dst[j] = src[j];
  }
}

// ---------------- K5-7: chunked parallel linear-recurrence scan ----------------
__global__ void k_scan_pass1(const float* __restrict__ dpT, const _Float16* __restrict__ uT,
                             const float* __restrict__ BmT, const float* Aneg,
                             float* __restrict__ Aprod, float* __restrict__ Hloc) {
  int bb = blockIdx.y, ch = blockIdx.x;
  int tid = threadIdx.x;           // 1024 = 64 d * 16 s
  int d = tid >> 4, s = tid & 15;
  float Ads = Aneg[d * 16 + s];
  size_t rowd = ((size_t)(bb * 64 + d)) * LSEQ + (size_t)ch * 128;
  const float* dpr = dpT + rowd;
  const _Float16* ur = uT + rowd;
  const float* Br = BmT + ((size_t)(bb * 16 + s)) * LSEQ + (size_t)ch * 128;
  float h = 0.f, ap = 1.f;
  for (int t = 0; t < 128; ++t) {
    float delta = softplus_f(dpr[t]);
    float dA = expf(delta * Ads);
    h = dA * h + delta * (float)ur[t] * Br[t];
    ap *= dA;
  }
  size_t o = ((size_t)(bb * 128 + ch)) * 1024 + tid;
  Aprod[o] = ap; Hloc[o] = h;
}

__global__ void k_scan_pass2(const float* __restrict__ Aprod, const float* __restrict__ Hloc,
                             float* __restrict__ Hin) {
  int bb = blockIdx.x, tid = threadIdx.x;
  float h = 0.f;
  for (int j = 0; j < 128; ++j) {
    size_t o = ((size_t)(bb * 128 + j)) * 1024 + tid;
    Hin[o] = h;
    h = Aprod[o] * h + Hloc[o];
  }
}

__global__ void k_scan_pass3(const float* __restrict__ dpT, const _Float16* __restrict__ uT,
                             const float* __restrict__ BmT, const float* __restrict__ CmT,
                             const _Float16* __restrict__ xzT, const float* Aneg, const float* Dp,
                             const float* __restrict__ Hin, _Float16* __restrict__ ygT) {
  __shared__ _Float16 ytile[64 * 128];
  int bb = blockIdx.y, ch = blockIdx.x;
  int tid = threadIdx.x;
  int d = tid >> 4, s = tid & 15;
  float Ads = Aneg[d * 16 + s];
  size_t rowd = ((size_t)(bb * 64 + d)) * LSEQ + (size_t)ch * 128;
  const float* dpr = dpT + rowd;
  const _Float16* ur = uT + rowd;
  const float* Br = BmT + ((size_t)(bb * 16 + s)) * LSEQ + (size_t)ch * 128;
  const float* Cr = CmT + ((size_t)(bb * 16 + s)) * LSEQ + (size_t)ch * 128;
  const _Float16* zr = xzT + ((size_t)(bb * 128 + 64 + d)) * LSEQ + (size_t)ch * 128;
  float Dd = Dp[d];
  float h = Hin[((size_t)(bb * 128 + ch)) * 1024 + tid];
  for (int t = 0; t < 128; ++t) {
    float delta = softplus_f(dpr[t]);
    float u = (float)ur[t];
    float dA = expf(delta * Ads);
    h = dA * h + delta * u * Br[t];
    float y = h * Cr[t];
    y += __shfl_xor(y, 1, 32);
    y += __shfl_xor(y, 2, 32);
    y += __shfl_xor(y, 4, 32);
    y += __shfl_xor(y, 8, 32);
    if (s == 0) {
      float z = (float)zr[t];
      ytile[d * 128 + t] = (_Float16)((y + u * Dd) * silu_f(z));
    }
  }
  __syncthreads();
  for (int e = tid; e < 64 * 128; e += 1024) {
    int dd = e >> 7, tt = e & 127;
    ygT[((size_t)(bb * 64 + dd)) * LSEQ + (size_t)ch * 128 + tt] = ytile[e];
  }
}

// ---------------- K8: ym = yg @ Wout^T + skip*xs, assembled into ycatT ----------------
__global__ void k_gemm_out_skip(const _Float16* __restrict__ ygT, const _Float16* __restrict__ B3,
                                const float* __restrict__ xnT, const float* skip,
                                float* __restrict__ ycatT) {
  __shared__ _Float16 At[16 * 64];
  __shared__ float Ot[32 * 16];
  int bb = blockIdx.y; int b = bb & 3, cidx = bb >> 2;
  int l0 = blockIdx.x * 16;
  int t = threadIdx.x;   // 64 threads = 2 waves
  for (int e = t; e < 1024; e += 64) {
    int k = e >> 4, m = e & 15;
    At[m * 64 + k] = ygT[((size_t)(bb * 64 + k)) * LSEQ + l0 + m];
  }
  __syncthreads();
  int w = t >> 5;
  int n0 = w * 16;
  v8f c = {};
#pragma unroll
  for (int ks = 0; ks < 2; ++ks) {
    v16h a = load_a_frag(At + ks * 32, 64);
    v16h bf = load_b_swz(B3, ks * 2 + w);
    c = __builtin_amdgcn_wmma_f32_16x16x32_f16(false, a, false, bf, (short)0, c, false, false);
  }
  { int lane = t & 31; int n = lane & 15, mb = (lane >> 4) << 3;
#pragma unroll
    for (int j = 0; j < 8; ++j) Ot[(n0 + n) * 16 + mb + j] = c[j]; }
  __syncthreads();
  float ss = *skip;
  if (t < 32) {
    size_t rowoff = ((size_t)(b * 128 + cidx * 32 + t)) * LSEQ + l0;
    const float* xrow = xnT + rowoff;
    float* dst = ycatT + rowoff;
    for (int j = 0; j < 16; ++j) dst[j] = Ot[t * 16 + j] + ss * xrow[j];
  }
}

// ---------------- K10: fused 5-way projection (K=128,N=128) + exact GELU ----------------
__global__ void k_gemm_proj(const _Float16* __restrict__ xmT, const _Float16* __restrict__ B4,
                            const float* bcat, float* __restrict__ out) {
  __shared__ _Float16 At[16 * 128];
  __shared__ float Ot[128 * 16];
  int row0 = blockIdx.x * 16;         // over (b,l) rows
  int b = row0 >> 14, l0 = row0 & (LSEQ - 1);
  int t = threadIdx.x;
  for (int e = t; e < 2048; e += 256) {
    int k = e >> 4, m = e & 15;
    At[m * 128 + k] = xmT[((size_t)(b * 128 + k)) * LSEQ + l0 + m];
  }
  __syncthreads();
  int w = t >> 5;
  int n0 = w * 16;
  v8f c = {};
#pragma unroll
  for (int ks = 0; ks < 4; ++ks) {
    v16h a = load_a_frag(At + ks * 32, 128);
    v16h bf = load_b_swz(B4, ks * 8 + w);
    c = __builtin_amdgcn_wmma_f32_16x16x32_f16(false, a, false, bf, (short)0, c, false, false);
  }
  { int lane = t & 31; int n = lane & 15, mb = (lane >> 4) << 3;
#pragma unroll
    for (int j = 0; j < 8; ++j) Ot[(n0 + n) * 16 + mb + j] = c[j]; }
  __syncthreads();
  if (t < 128) {
    int e = t, cp, coff, cnt;
    if (e < 8)       { cp = e;      coff = 0;  cnt = 8; }
    else if (e < 24) { cp = e - 8;  coff = 8;  cnt = 16; }
    else if (e < 48) { cp = e - 24; coff = 24; cnt = 24; }
    else if (e < 80) { cp = e - 48; coff = 48; cnt = 32; }
    else             { cp = e - 80; coff = 80; cnt = 48; }
    size_t base = (size_t)4 * LSEQ * coff + ((size_t)(b * cnt + cp)) * LSEQ + l0;
    float bias = bcat[e];
    for (int j = 0; j < 16; ++j) {
      float v = Ot[e * 16 + j] + bias;
      out[base + j] = 0.5f * v * (1.0f + erff(v * 0.70710678118654752f));
    }
  }
}

// ---------------- launcher ----------------
extern "C" void kernel_launch(void* const* d_in, const int* in_sizes, int n_in,
                              void* d_out, int out_size, void* d_ws, size_t ws_size,
                              hipStream_t stream) {
  (void)in_sizes; (void)n_in; (void)out_size; (void)ws_size;
  const float* x     = (const float*)d_in[0];
  const float* ln_g  = (const float*)d_in[1];
  const float* ln_b  = (const float*)d_in[2];
  const float* skip  = (const float*)d_in[3];
  const float* Win   = (const float*)d_in[4];
  const float* Wconv = (const float*)d_in[5];
  const float* bconv = (const float*)d_in[6];
  const float* Wx    = (const float*)d_in[7];
  const float* Wdt   = (const float*)d_in[8];
  const float* bdt   = (const float*)d_in[9];
  const float* A_log = (const float*)d_in[10];
  const float* Dp    = (const float*)d_in[11];
  const float* Wout  = (const float*)d_in[12];
  const float* W1 = (const float*)d_in[13]; const float* b1 = (const float*)d_in[14];
  const float* W2 = (const float*)d_in[15]; const float* b2 = (const float*)d_in[16];
  const float* W3 = (const float*)d_in[17]; const float* b3 = (const float*)d_in[18];
  const float* W4 = (const float*)d_in[19]; const float* b4 = (const float*)d_in[20];
  const float* W5 = (const float*)d_in[21]; const float* b5 = (const float*)d_in[22];

  char* ws = (char*)d_ws;
  float*     xnT   = (float*)    (ws + OFF_XNT);
  _Float16*  xzT   = (_Float16*) (ws + OFF_XZT);
  _Float16*  uT    = (_Float16*) (ws + OFF_UT);
  float*     dpT   = (float*)    (ws + OFF_DPT);
  float*     BmT   = (float*)    (ws + OFF_BMT);
  float*     CmT   = (float*)    (ws + OFF_CMT);
  float*     Aprod = (float*)    (ws + OFF_APROD);
  float*     Hloc  = (float*)    (ws + OFF_HLOC);
  float*     Hin   = (float*)    (ws + OFF_HIN);
  _Float16*  ygT   = (_Float16*) (ws + OFF_YGT);
  float*     ycatT = (float*)    (ws + OFF_YCATT);
  _Float16*  xmT   = (_Float16*) (ws + OFF_XMT);
  _Float16*  B1    = (_Float16*) (ws + OFF_B1);
  _Float16*  B2    = (_Float16*) (ws + OFF_B2);
  _Float16*  B3    = (_Float16*) (ws + OFF_B3);
  _Float16*  B4    = (_Float16*) (ws + OFF_B4);
  float*     bcat  = (float*)    (ws + OFF_BCAT);
  float*     Aneg  = (float*)    (ws + OFF_ANEG);

  k_prep<<<1, 256, 0, stream>>>(Win, Wx, Wdt, A_log, W1, b1, W2, b2, W3, b3, W4, b4, W5, b5,
                                Wout, B1, B2, B3, B4, bcat, Aneg);
  k_layernorm<float><<<256, 256, 0, stream>>>(x, ln_g, ln_b, xnT);
  k_gemm_xz<<<dim3(1024, 16), 256, 0, stream>>>(xnT, B1, xzT);
  k_conv<<<65536, 256, 0, stream>>>(xzT, Wconv, bconv, uT);
  k_gemm_xdbl<<<dim3(1024, 16), 192, 0, stream>>>(uT, B2, bdt, dpT, BmT, CmT);
  k_scan_pass1<<<dim3(128, 16), 1024, 0, stream>>>(dpT, uT, BmT, Aneg, Aprod, Hloc);
  k_scan_pass2<<<16, 1024, 0, stream>>>(Aprod, Hloc, Hin);
  k_scan_pass3<<<dim3(128, 16), 1024, 0, stream>>>(dpT, uT, BmT, CmT, xzT, Aneg, Dp, Hin, ygT);
  k_gemm_out_skip<<<dim3(1024, 16), 64, 0, stream>>>(ygT, B3, xnT, skip, ycatT);
  k_layernorm<_Float16><<<256, 256, 0, stream>>>(ycatT, ln_g, ln_b, xmT);
  k_gemm_proj<<<4096, 256, 0, stream>>>(xmT, B4, bcat, (float*)d_out);
}
